// GPS_16870631539209
// MI455X (gfx1250) — matrix operand order repaired and checked by reading, not compiled, not run on hardware
//
#include <hip/hip_runtime.h>

// ---------------------------------------------------------------------------
// Types for CDNA5 WMMA (wave32, v_wmma_f32_16x16x32_bf16)
// ---------------------------------------------------------------------------
typedef __bf16 bf16;
typedef __attribute__((ext_vector_type(16))) __bf16 v16bf;
typedef __attribute__((ext_vector_type(8)))  __bf16 v8bf;
typedef __attribute__((ext_vector_type(8)))  float   v8f;

#define Nn    65536
#define Ee    524288
#define Gg    256
#define Cch   256
#define Llay  4
#define Hhead 4
#define INDIM 771
#define KPAD  800   // node-encoder K padded to a multiple of 32

__device__ __forceinline__ v8f wmma_bf(v16bf a, v16bf b, v8f c) {
  return __builtin_amdgcn_wmma_f32_16x16x32_bf16(false, a, false, b, (short)0, c,
                                                 false, false);
}

// A fragment: 16x32 bf16 tile from LDS, row-major, row stride = STRIDE elems.
// lanes 0-15 -> rows 0-15, K {0..7,16..23}; lanes 16-31 -> K {8..15,24..31}.
template <int STRIDE>
__device__ __forceinline__ v16bf load_a_frag(const bf16* base, int lane) {
  int half = lane >> 4, r = lane & 15;
  const bf16* p = base + r * STRIDE + half * 8;
  v8bf lo = *(const v8bf*)(p);
  v8bf hi = *(const v8bf*)(p + 16);
  v16bf out;
#pragma unroll
  for (int i = 0; i < 8; ++i) { out[i] = lo[i]; out[i + 8] = hi[i]; }
  return out;
}

// B fragment: 32x16 bf16 tile; LDS holds B transposed ([col][k], stride=STRIDE).
// lanes 0-15 -> col=lane, K {0..7,16..23}; lanes 16-31 -> col=lane-16, K {8..15,24..31}.
template <int STRIDE>
__device__ __forceinline__ v16bf load_b_frag(const bf16* base, int lane) {
  int half = lane >> 4, c = lane & 15;
  const bf16* p = base + c * STRIDE + half * 8;
  v8bf lo = *(const v8bf*)(p);
  v8bf hi = *(const v8bf*)(p + 16);
  v16bf out;
#pragma unroll
  for (int i = 0; i < 8; ++i) { out[i] = lo[i]; out[i + 8] = hi[i]; }
  return out;
}

__device__ __forceinline__ float wave_sum(float v) {
#pragma unroll
  for (int o = 16; o; o >>= 1) v += __shfl_xor(v, o, 32);
  return v;
}
__device__ __forceinline__ float wave_max(float v) {
#pragma unroll
  for (int o = 16; o; o >>= 1) v = fmaxf(v, __shfl_xor(v, o, 32));
  return v;
}

// ---------------------------------------------------------------------------
// Generic bf16 GEMM: C = A[MxK] * B[KxN] + bias, templated branch-free epilogue
// block tile 128x128, 8 waves, wave tile 32x64 (2x4 WMMA accumulators)
// M, N multiples of 128; K multiple of 32 (branch-free staging).
// ---------------------------------------------------------------------------
template <bool HAS_CF, bool HAS_CB, bool RELU>
__global__ __launch_bounds__(256) void gemm_kernel(
    const bf16* __restrict__ A, int lda, const bf16* __restrict__ B, int ldb,
    float* __restrict__ Cf, bf16* __restrict__ Cb, int ldc,
    const float* __restrict__ bias, int M, int N, int K) {
  __shared__ bf16 sA[128][32];
  __shared__ bf16 sB[128][32];  // B transposed: [col][k]
  int bm0 = blockIdx.y * 128, bn0 = blockIdx.x * 128;
  int tid = threadIdx.x, lane = tid & 31, w = tid >> 5;
  int wm = w & 3, wn = w >> 2;
  int arow = tid >> 1, ac0 = (tid & 1) * 16;
  int bcol = tid & 127, bkh = (tid >> 7) * 16;
  v8f acc[2][4];
#pragma unroll
  for (int mi = 0; mi < 2; ++mi)
#pragma unroll
    for (int ni = 0; ni < 4; ++ni)
#pragma unroll
      for (int j = 0; j < 8; ++j) acc[mi][ni][j] = 0.f;

  for (int k0 = 0; k0 < K; k0 += 32) {
    // stage A: 16 contiguous bf16 per thread -> 2 x b128 each way
    {
      const bf16* g = A + (size_t)(bm0 + arow) * lda + k0 + ac0;
      v8bf t0 = *(const v8bf*)g;
      v8bf t1 = *(const v8bf*)(g + 8);
      *(v8bf*)&sA[arow][ac0] = t0;
      *(v8bf*)&sA[arow][ac0 + 8] = t1;
    }
    // stage B transposed: gather 16 rows (coalesced across threads), b128 out
    {
      const bf16* g = B + (size_t)(k0 + bkh) * ldb + bn0 + bcol;
      v8bf t0, t1;
#pragma unroll
      for (int j = 0; j < 8; ++j) t0[j] = g[(size_t)j * ldb];
#pragma unroll
      for (int j = 0; j < 8; ++j) t1[j] = g[(size_t)(j + 8) * ldb];
      *(v8bf*)&sB[bcol][bkh] = t0;
      *(v8bf*)&sB[bcol][bkh + 8] = t1;
    }
    __syncthreads();
    v16bf af[2], bfg[4];
#pragma unroll
    for (int mi = 0; mi < 2; ++mi)
      af[mi] = load_a_frag<32>(&sA[wm * 32 + mi * 16][0], lane);
#pragma unroll
    for (int ni = 0; ni < 4; ++ni)
      bfg[ni] = load_b_frag<32>(&sB[wn * 64 + ni * 16][0], lane);
#pragma unroll
    for (int mi = 0; mi < 2; ++mi)
#pragma unroll
      for (int ni = 0; ni < 4; ++ni)
        acc[mi][ni] = wmma_bf(af[mi], bfg[ni], acc[mi][ni]);
    __syncthreads();
  }
  int lr = (lane >> 4) * 8, lc = lane & 15;
#pragma unroll
  for (int mi = 0; mi < 2; ++mi)
#pragma unroll
    for (int ni = 0; ni < 4; ++ni) {
      int r0 = bm0 + wm * 32 + mi * 16 + lr;
      int c = bn0 + wn * 64 + ni * 16 + lc;
      float bv = bias[c];
#pragma unroll
      for (int j = 0; j < 8; ++j) {
        float v = acc[mi][ni][j] + bv;
        if (RELU) v = fmaxf(v, 0.f);
        size_t o = (size_t)(r0 + j) * ldc + c;
        if (HAS_CF) Cf[o] = v;
        if (HAS_CB) Cb[o] = (bf16)v;
      }
    }
}

// ---------------------------------------------------------------------------
// Attention: scores S[gh][q][k] = (Q . K^T) * 1/sqrt(DH); q,k from fp32 qkv
// ---------------------------------------------------------------------------
__global__ __launch_bounds__(256) void attn_scores_kernel(
    const float* __restrict__ qkv, float* __restrict__ S) {
  int gh = blockIdx.z, g = gh >> 2, hd = gh & 3;
  int bm0 = blockIdx.y * 128, bn0 = blockIdx.x * 128;
  __shared__ bf16 sA[128][64];
  __shared__ bf16 sB[128][64];
  int tid = threadIdx.x;
  const float* qbase = qkv + (size_t)g * 256 * 768 + hd * 64;
  const float* kbase = qbase + 256;
  {
    int row = tid >> 1, c0 = (tid & 1) * 32;
    const float4* p = (const float4*)(qbase + (size_t)(bm0 + row) * 768 + c0);
    const float4* p2 = (const float4*)(kbase + (size_t)(bn0 + row) * 768 + c0);
#pragma unroll
    for (int u = 0; u < 4; ++u) {
      float4 fa = p[u * 2], fb = p[u * 2 + 1];
      float4 ga = p2[u * 2], gb2 = p2[u * 2 + 1];
      v8bf ta, tb;
      ta[0] = (bf16)fa.x; ta[1] = (bf16)fa.y; ta[2] = (bf16)fa.z; ta[3] = (bf16)fa.w;
      ta[4] = (bf16)fb.x; ta[5] = (bf16)fb.y; ta[6] = (bf16)fb.z; ta[7] = (bf16)fb.w;
      tb[0] = (bf16)ga.x; tb[1] = (bf16)ga.y; tb[2] = (bf16)ga.z; tb[3] = (bf16)ga.w;
      tb[4] = (bf16)gb2.x; tb[5] = (bf16)gb2.y; tb[6] = (bf16)gb2.z; tb[7] = (bf16)gb2.w;
      *(v8bf*)&sA[row][c0 + u * 8] = ta;
      *(v8bf*)&sB[row][c0 + u * 8] = tb;
    }
  }
  __syncthreads();
  int lane = tid & 31, w = tid >> 5, wm = w & 3, wn = w >> 2;
  v8f acc[2][4];
#pragma unroll
  for (int mi = 0; mi < 2; ++mi)
#pragma unroll
    for (int ni = 0; ni < 4; ++ni)
#pragma unroll
      for (int j = 0; j < 8; ++j) acc[mi][ni][j] = 0.f;
#pragma unroll
  for (int k0 = 0; k0 < 64; k0 += 32) {
    v16bf a0 = load_a_frag<64>(&sA[wm * 32][k0], lane);
    v16bf a1 = load_a_frag<64>(&sA[wm * 32 + 16][k0], lane);
#pragma unroll
    for (int ni = 0; ni < 4; ++ni) {
      v16bf bfg = load_b_frag<64>(&sB[wn * 64 + ni * 16][k0], lane);
      acc[0][ni] = wmma_bf(a0, bfg, acc[0][ni]);
      acc[1][ni] = wmma_bf(a1, bfg, acc[1][ni]);
    }
  }
  float* Sg = S + (size_t)gh * 65536;
  int lr = (lane >> 4) * 8, lc = lane & 15;
#pragma unroll
  for (int mi = 0; mi < 2; ++mi)
#pragma unroll
    for (int ni = 0; ni < 4; ++ni)
#pragma unroll
      for (int j = 0; j < 8; ++j) {
        int r = bm0 + wm * 32 + mi * 16 + lr + j;
        int c = bn0 + wn * 64 + ni * 16 + lc;
        Sg[(size_t)r * 256 + c] = acc[mi][ni][j] * 0.125f;
      }
}

__global__ __launch_bounds__(256) void softmax_kernel(
    const float* __restrict__ S, bf16* __restrict__ P) {
  int row = blockIdx.x * 8 + (threadIdx.x >> 5);
  int lane = threadIdx.x & 31;
  const float* x = S + (size_t)row * 256;
  float v[8], m = -1e30f;
#pragma unroll
  for (int j = 0; j < 8; ++j) { v[j] = x[j * 32 + lane]; m = fmaxf(m, v[j]); }
  m = wave_max(m);
  float s = 0.f;
#pragma unroll
  for (int j = 0; j < 8; ++j) { v[j] = __expf(v[j] - m); s += v[j]; }
  s = wave_sum(s);
  float inv = 1.f / s;
  bf16* y = P + (size_t)row * 256;
#pragma unroll
  for (int j = 0; j < 8; ++j) y[j * 32 + lane] = (bf16)(v[j] * inv);
}

// O[g*256+q][hd*64+d] = sum_k P[gh][q][k] * V[gh][k][d]
__global__ __launch_bounds__(256) void attn_pv_kernel(
    const bf16* __restrict__ P, const float* __restrict__ qkv,
    float* __restrict__ O) {
  int gh = blockIdx.y, g = gh >> 2, hd = gh & 3;
  int bm0 = blockIdx.x * 128;
  __shared__ bf16 sA[128][32];
  __shared__ bf16 sB[64][32];
  int tid = threadIdx.x, lane = tid & 31, w = tid >> 5;
  int wm = w & 3, wn = w >> 2;
  const bf16* Pg = P + (size_t)gh * 65536;
  const float* vbase = qkv + (size_t)g * 256 * 768 + 512 + hd * 64;
  v8f acc[2][2];
#pragma unroll
  for (int mi = 0; mi < 2; ++mi)
#pragma unroll
    for (int ni = 0; ni < 2; ++ni)
#pragma unroll
      for (int j = 0; j < 8; ++j) acc[mi][ni][j] = 0.f;
  int arow = tid >> 1, ac0 = (tid & 1) * 16;
  int bn = tid & 63, bkb = (tid >> 6) * 8;
  for (int k0 = 0; k0 < 256; k0 += 32) {
    {
      const bf16* p = Pg + (size_t)(bm0 + arow) * 256 + k0 + ac0;
      v8bf t0 = *(const v8bf*)p;
      v8bf t1 = *(const v8bf*)(p + 8);
      *(v8bf*)&sA[arow][ac0] = t0;
      *(v8bf*)&sA[arow][ac0 + 8] = t1;
    }
    {
      const float* gv = vbase + (size_t)(k0 + bkb) * 768 + bn;
      v8bf t;
#pragma unroll
      for (int j = 0; j < 8; ++j) t[j] = (bf16)gv[(size_t)j * 768];
      *(v8bf*)&sB[bn][bkb] = t;
    }
    __syncthreads();
    v16bf a0 = load_a_frag<32>(&sA[wm * 32][0], lane);
    v16bf a1 = load_a_frag<32>(&sA[wm * 32 + 16][0], lane);
    v16bf b0 = load_b_frag<32>(&sB[wn * 32][0], lane);
    v16bf b1 = load_b_frag<32>(&sB[wn * 32 + 16][0], lane);
    acc[0][0] = wmma_bf(a0, b0, acc[0][0]);
    acc[0][1] = wmma_bf(a0, b1, acc[0][1]);
    acc[1][0] = wmma_bf(a1, b0, acc[1][0]);
    acc[1][1] = wmma_bf(a1, b1, acc[1][1]);
    __syncthreads();
  }
  int lr = (lane >> 4) * 8, lc = lane & 15;
#pragma unroll
  for (int mi = 0; mi < 2; ++mi)
#pragma unroll
    for (int ni = 0; ni < 2; ++ni)
#pragma unroll
      for (int j = 0; j < 8; ++j) {
        int row = g * 256 + bm0 + wm * 32 + mi * 16 + lr + j;
        int col = hd * 64 + wn * 32 + ni * 16 + lc;
        O[(size_t)row * 256 + col] = acc[mi][ni][j];
      }
}

// ---------------------------------------------------------------------------
// Elementwise / reduction kernels
// ---------------------------------------------------------------------------
__global__ __launch_bounds__(256) void cvt_kernel(const float* __restrict__ x,
                                                  bf16* __restrict__ y,
                                                  size_t n) {
  size_t i = (size_t)blockIdx.x * 256 + threadIdx.x;
  if (i < n) y[i] = (bf16)x[i];
}

// pad-convert x [N,771] f32 -> [N,KPAD] bf16 (zero-filled tail)
__global__ __launch_bounds__(256) void pad_x_kernel(const float* __restrict__ x,
                                                    bf16* __restrict__ y) {
  size_t i = (size_t)blockIdx.x * 256 + threadIdx.x;
  int k = (int)(i % KPAD);
  size_t r = i / KPAD;
  y[i] = (k < INDIM) ? (bf16)x[r * INDIM + k] : (bf16)0.f;
}

// pad-convert ne_w [771,256] f32 -> [KPAD,256] bf16 (zero-filled tail rows)
__global__ __launch_bounds__(256) void pad_w_kernel(const float* __restrict__ w,
                                                    bf16* __restrict__ y) {
  size_t i = (size_t)blockIdx.x * 256 + threadIdx.x;  // over KPAD*256
  int k = (int)(i >> 8);
  y[i] = (k < INDIM) ? (bf16)w[i] : (bf16)0.f;
}

// dst[l][c][r] = src[l][r][c]   (for W used as x @ W^T)
__global__ __launch_bounds__(256) void cvt_tr_kernel(
    const float* __restrict__ s, bf16* __restrict__ d, int Lb, int R, int Cw) {
  size_t i = (size_t)blockIdx.x * 256 + threadIdx.x;
  size_t tot = (size_t)Lb * R * Cw;
  if (i >= tot) return;
  int l = (int)(i / ((size_t)R * Cw));
  size_t rem = i - (size_t)l * R * Cw;
  int r = (int)(rem / Cw), c = (int)(rem % Cw);
  d[(size_t)l * R * Cw + (size_t)c * R + r] = (bf16)s[i];
}

__global__ __launch_bounds__(256) void zero_kernel(float* __restrict__ x,
                                                   size_t n) {
  size_t i = (size_t)blockIdx.x * 256 + threadIdx.x;
  if (i < n) x[i] = 0.f;
}

__global__ __launch_bounds__(256) void edge_enc_kernel(
    const float* __restrict__ ea, const float* __restrict__ w,
    const float* __restrict__ b, bf16* __restrict__ out) {
  size_t i = (size_t)blockIdx.x * 256 + threadIdx.x;
  int c = (int)(i & 255);
  size_t e = i >> 8;
  float s = b[c];
#pragma unroll
  for (int k = 0; k < 4; ++k) s += ea[e * 4 + k] * w[k * 256 + c];
  out[i] = (bf16)fmaxf(s, 0.f);
}

__global__ __launch_bounds__(256) void gine_scatter_kernel(
    const float* __restrict__ ee, const float* __restrict__ h,
    const int* __restrict__ src, const int* __restrict__ dst,
    float* __restrict__ aggr) {
  size_t i = (size_t)blockIdx.x * 256 + threadIdx.x;
  int c = (int)(i & 255);
  size_t e = i >> 8;
  float v = fmaxf(h[(size_t)src[e] * 256 + c] + ee[i], 0.f);
  atomicAdd(&aggr[(size_t)dst[e] * 256 + c], v);
}

__global__ __launch_bounds__(256) void gine_comb_kernel(
    const float* __restrict__ h, const float* __restrict__ aggr,
    const float* __restrict__ epsp, int l, bf16* __restrict__ out, size_t n) {
  size_t i = (size_t)blockIdx.x * 256 + threadIdx.x;
  if (i >= n) return;
  float ev = 1.f + epsp[l];
  out[i] = (bf16)(ev * h[i] + aggr[i]);
}

__global__ __launch_bounds__(256) void add2_kernel(const float* __restrict__ a,
                                                   const float* __restrict__ b,
                                                   float* __restrict__ yf,
                                                   bf16* __restrict__ yb,
                                                   size_t n) {
  size_t i = (size_t)blockIdx.x * 256 + threadIdx.x;
  if (i >= n) return;
  float v = a[i] + b[i];
  yf[i] = v;
  yb[i] = (bf16)v;
}

// LayerNorm over C=256; 1 wave per row. y = LN(x + r) * g + b (opt relu).
__global__ __launch_bounds__(256) void ln_kernel(
    const float* __restrict__ X, const float* __restrict__ R,
    const float* __restrict__ gam, const float* __restrict__ bet,
    float* __restrict__ Yf, bf16* __restrict__ Yb, int relu) {
  int row = blockIdx.x * 8 + (threadIdx.x >> 5);
  int lane = threadIdx.x & 31;
  const float* x = X + (size_t)row * 256;
  float v[8], s = 0.f;
#pragma unroll
  for (int j = 0; j < 8; ++j) {
    int c = j * 32 + lane;
    float t = x[c];
    if (R) t += R[(size_t)row * 256 + c];
    v[j] = t;
    s += t;
  }
  float mean = wave_sum(s) * (1.f / 256.f);
  float q = 0.f;
#pragma unroll
  for (int j = 0; j < 8; ++j) { float d = v[j] - mean; q += d * d; }
  float inv = rsqrtf(wave_sum(q) * (1.f / 256.f) + 1e-5f);
#pragma unroll
  for (int j = 0; j < 8; ++j) {
    int c = j * 32 + lane;
    float y = (v[j] - mean) * inv * gam[c] + bet[c];
    if (relu) y = fmaxf(y, 0.f);
    size_t o = (size_t)row * 256 + c;
    if (Yf) Yf[o] = y;
    if (Yb) Yb[o] = (bf16)y;
  }
}

// BatchNorm stats over rows, one block per channel. mv[0:256]=mean, [256:512]=var
__global__ __launch_bounds__(256) void bn_stats_kernel(
    const float* __restrict__ X, int rows, float* __restrict__ mv) {
  int ch = blockIdx.x, tid = threadIdx.x;
  float s = 0.f, q = 0.f;
  for (int i = tid; i < rows; i += 256) {
    float v = X[(size_t)i * 256 + ch];
    s += v;
    q += v * v;
  }
  __shared__ float ss[256], sq[256];
  ss[tid] = s;
  sq[tid] = q;
  __syncthreads();
  for (int o = 128; o > 0; o >>= 1) {
    if (tid < o) { ss[tid] += ss[tid + o]; sq[tid] += sq[tid + o]; }
    __syncthreads();
  }
  if (tid == 0) {
    float m = ss[0] / rows;
    mv[ch] = m;
    mv[256 + ch] = sq[0] / rows - m * m;
  }
}

__global__ __launch_bounds__(256) void bn_apply_kernel(
    const float* __restrict__ X, const float* __restrict__ mv,
    const float* __restrict__ g, const float* __restrict__ b,
    bf16* __restrict__ Y, size_t n) {
  size_t i = (size_t)blockIdx.x * 256 + threadIdx.x;
  if (i >= n) return;
  int c = (int)(i & 255);
  float y = (X[i] - mv[c]) * rsqrtf(mv[256 + c] + 1e-5f) * g[c] + b[c];
  Y[i] = (bf16)fmaxf(y, 0.f);
}

__global__ __launch_bounds__(256) void pool_kernel(const float* __restrict__ h,
                                                   float* __restrict__ gf,
                                                   bf16* __restrict__ gb) {
  int g = blockIdx.x, c = threadIdx.x;
  float s = 0.f;
  for (int i = 0; i < 256; ++i) s += h[((size_t)g * 256 + i) * 256 + c];
  float m = s * (1.f / 256.f);
  gf[(size_t)g * 256 + c] = m;
  gb[(size_t)g * 256 + c] = (bf16)m;
}

__global__ __launch_bounds__(256) void cls_final_kernel(
    const bf16* __restrict__ z, const float* __restrict__ w,
    const float* __restrict__ b, float* __restrict__ out) {
  int g = threadIdx.x;
  float s = b[0];
#pragma unroll 4
  for (int k = 0; k < 128; ++k) s += (float)z[(size_t)g * 128 + k] * w[k];
  out[g] = s;
}

// ---------------------------------------------------------------------------
// Host orchestration
// ---------------------------------------------------------------------------
extern "C" void kernel_launch(void* const* d_in, const int* in_sizes, int n_in,
                              void* d_out, int out_size, void* d_ws,
                              size_t ws_size, hipStream_t stream) {
  const float* x         = (const float*)d_in[0];
  const int*   eidx      = (const int*)d_in[1];
  const float* edge_attr = (const float*)d_in[3];
  const float* ne_w      = (const float*)d_in[4];
  const float* ne_b      = (const float*)d_in[5];
  const float* ne_ln_g   = (const float*)d_in[6];
  const float* ne_ln_b   = (const float*)d_in[7];
  const float* ee_w      = (const float*)d_in[8];
  const float* ee_b      = (const float*)d_in[9];
  const float* epsv      = (const float*)d_in[10];
  const float* elin_w    = (const float*)d_in[11];
  const float* gin_w1    = (const float*)d_in[12];
  const float* gin_w2    = (const float*)d_in[13];
  const float* attn_in_w = (const float*)d_in[14];
  const float* attn_out_w= (const float*)d_in[15];
  const float* mlp_w1    = (const float*)d_in[16];
  const float* mlp_w2    = (const float*)d_in[17];
  const float* elin_b    = (const float*)d_in[18];
  const float* gin_b1    = (const float*)d_in[19];
  const float* gin_b2    = (const float*)d_in[20];
  const float* attn_in_b = (const float*)d_in[21];
  const float* attn_out_b= (const float*)d_in[22];
  const float* mlp_b1    = (const float*)d_in[23];
  const float* mlp_b2    = (const float*)d_in[24];
  const float* gin_bn_g  = (const float*)d_in[25];
  const float* ln1_g     = (const float*)d_in[26];
  const float* ln2_g     = (const float*)d_in[27];
  const float* ln3_g     = (const float*)d_in[28];
  const float* gin_bn_b  = (const float*)d_in[29];
  const float* ln1_b     = (const float*)d_in[30];
  const float* ln2_b     = (const float*)d_in[31];
  const float* ln3_b     = (const float*)d_in[32];
  const float* cls_w1    = (const float*)d_in[33];
  const float* cls_b1    = (const float*)d_in[34];
  const float* cls_ln_g  = (const float*)d_in[35];
  const float* cls_ln_b  = (const float*)d_in[36];
  const float* cls_w2    = (const float*)d_in[37];
  const float* cls_b2    = (const float*)d_in[38];
  const float* cls_w3    = (const float*)d_in[39];
  const float* cls_b3    = (const float*)d_in[40];

  const int* src = eidx;
  const int* dst = eidx + Ee;

  const size_t NC = (size_t)Nn * Cch;   // 16,777,216
  const size_t EC = (size_t)Ee * Cch;   // 134,217,728

  // bf16 weight arena offsets (elements)
  const size_t OFF_NEW  = 0;                         // KPAD*256 (padded)
  const size_t OFF_ELIN = OFF_NEW + (size_t)KPAD * 256;
  const size_t OFF_GIN1 = OFF_ELIN + 4 * 65536;
  const size_t OFF_GIN2 = OFF_GIN1 + 4 * 65536;
  const size_t OFF_AIN  = OFF_GIN2 + 4 * 65536;      // 4*768*256 (transposed)
  const size_t OFF_AOUT = OFF_AIN + 4 * 196608;      // 4*256*256 (transposed)
  const size_t OFF_M1   = OFF_AOUT + 4 * 65536;      // 4*256*512
  const size_t OFF_M2   = OFF_M1 + 4 * 131072;       // 4*512*256
  const size_t OFF_C1   = OFF_M2 + 4 * 131072;       // 256*256
  const size_t OFF_C2   = OFF_C1 + 65536;            // 256*128
  const size_t WB_ELEMS = OFF_C2 + 32768;

  char* ws = (char*)d_ws;
  size_t off = 0;
  auto carve = [&](size_t bytes) -> void* {
    void* p = ws + off;
    off += (bytes + 255) & ~(size_t)255;
    return p;
  };
  bf16*  WB   = (bf16*)carve(WB_ELEMS * 2);
  float* Hbuf = (float*)carve(NC * 4);
  bf16*  HB   = (bf16*)carve(NC * 2);
  bf16*  EB   = (bf16*)carve(EC * 2);
  float* T1   = (float*)carve(EC * 4);      // ee / qkv(+scores) / x staging
  float* T2   = (float*)carve(NC * 4);
  float* T3   = (float*)carve(NC * 4);
  float* T4   = (float*)carve(NC * 4);
  bf16*  TBB  = (bf16*)carve((size_t)Nn * 512 * 2 * 2);  // TB1|TB2 / P
  float* MV   = (float*)carve(512 * 4);
  float* GEMB = (float*)carve((size_t)Gg * Cch * 4);
  bf16*  GEMBB= (bf16*)carve((size_t)Gg * Cch * 2);
  float* Z1   = (float*)carve((size_t)Gg * Cch * 4);
  bf16*  Z1B  = (bf16*)carve((size_t)Gg * Cch * 2);
  bf16*  Z2B  = (bf16*)carve((size_t)Gg * 128 * 2);

  bf16* TB1 = TBB;
  bf16* TB2 = TBB + (size_t)Nn * 512;
  bf16* Pbuf = TBB;                               // 128 MiB (spans TB1+TB2)
  float* QKV = T1;                                // N*768 f32 = 192 MiB
  float* Sbuf = (float*)((char*)T1 + ((size_t)1 << 28));  // T1 + 256 MiB
  bf16* XB = (bf16*)T1;                           // x bf16 [N,KPAD] staging

  auto cvt = [&](const float* s, bf16* d, size_t n) {
    cvt_kernel<<<(unsigned)((n + 255) / 256), 256, 0, stream>>>(s, d, n);
  };
  // f32-output GEMM (bias, no relu)
  auto gemm_f = [&](const bf16* A, int lda, const bf16* B, int ldb, float* Cf,
                    int ldc, const float* bias, int M, int N, int K) {
    dim3 grid(N / 128, M / 128);
    gemm_kernel<true, false, false><<<grid, 256, 0, stream>>>(
        A, lda, B, ldb, Cf, nullptr, ldc, bias, M, N, K);
  };
  // bf16-output GEMM with ReLU (bias)
  auto gemm_br = [&](const bf16* A, int lda, const bf16* B, int ldb, bf16* Cb,
                     int ldc, const float* bias, int M, int N, int K) {
    dim3 grid(N / 128, M / 128);
    gemm_kernel<false, true, true><<<grid, 256, 0, stream>>>(
        A, lda, B, ldb, nullptr, Cb, ldc, bias, M, N, K);
  };

  // ---- weight conversion (fp32 -> bf16, attn weights transposed) ----
  pad_w_kernel<<<(KPAD * 256) / 256, 256, 0, stream>>>(ne_w, WB + OFF_NEW);
  cvt(elin_w, WB + OFF_ELIN, (size_t)4 * 65536);
  cvt(gin_w1, WB + OFF_GIN1, (size_t)4 * 65536);
  cvt(gin_w2, WB + OFF_GIN2, (size_t)4 * 65536);
  cvt_tr_kernel<<<(4 * 768 * 256 + 255) / 256, 256, 0, stream>>>(
      attn_in_w, WB + OFF_AIN, 4, 768, 256);
  cvt_tr_kernel<<<(4 * 256 * 256 + 255) / 256, 256, 0, stream>>>(
      attn_out_w, WB + OFF_AOUT, 4, 256, 256);
  cvt(mlp_w1, WB + OFF_M1, (size_t)4 * 131072);
  cvt(mlp_w2, WB + OFF_M2, (size_t)4 * 131072);
  cvt(cls_w1, WB + OFF_C1, 65536);
  cvt(cls_w2, WB + OFF_C2, 32768);

  // ---- encoders ----
  pad_x_kernel<<<(unsigned)(((size_t)Nn * KPAD) / 256), 256, 0, stream>>>(x, XB);
  gemm_f(XB, KPAD, WB + OFF_NEW, 256, T2, 256, ne_b, Nn, 256, KPAD);
  ln_kernel<<<Nn / 8, 256, 0, stream>>>(T2, nullptr, ne_ln_g, ne_ln_b, Hbuf, HB,
                                        1);
  edge_enc_kernel<<<(unsigned)(EC / 256), 256, 0, stream>>>(edge_attr, ee_w,
                                                            ee_b, EB);

  // ---- GPS layers ----
  for (int l = 0; l < Llay; ++l) {
    // GINEConv edge transform: ee = e @ elin_w[l] + b   (E x 256)
    gemm_f(EB, 256, WB + OFF_ELIN + (size_t)l * 65536, 256, T1, 256,
           elin_b + l * 256, Ee, 256, 256);
    // aggr = segment_sum(relu(h[src] + ee), dst)
    zero_kernel<<<(unsigned)(NC / 256), 256, 0, stream>>>(T2, NC);
    gine_scatter_kernel<<<(unsigned)(EC / 256), 256, 0, stream>>>(T1, Hbuf, src,
                                                                  dst, T2);
    // g1 = (1+eps)h + aggr -> bf16
    gine_comb_kernel<<<(unsigned)(NC / 256), 256, 0, stream>>>(Hbuf, T2, epsv,
                                                               l, TB1, NC);
    gemm_f(TB1, 256, WB + OFF_GIN1 + (size_t)l * 65536, 256, T3, 256,
           gin_b1 + l * 256, Nn, 256, 256);
    bn_stats_kernel<<<256, 256, 0, stream>>>(T3, Nn, MV);
    bn_apply_kernel<<<(unsigned)(NC / 256), 256, 0, stream>>>(
        T3, MV, gin_bn_g + l * 256, gin_bn_b + l * 256, TB1, NC);
    gemm_f(TB1, 256, WB + OFF_GIN2 + (size_t)l * 65536, 256, T4, 256,
           gin_b2 + l * 256, Nn, 256, 256);
    // h_local = LN(gin_out + h)
    ln_kernel<<<Nn / 8, 256, 0, stream>>>(T4, Hbuf, ln1_g + l * 256,
                                          ln1_b + l * 256, T4, nullptr, 0);

    // ---- global attention ----
    gemm_f(HB, 256, WB + OFF_AIN + (size_t)l * 196608, 768, QKV, 768,
           attn_in_b + l * 768, Nn, 768, 256);
    {
      dim3 grid(2, 2, Gg * Hhead);
      attn_scores_kernel<<<grid, 256, 0, stream>>>(QKV, Sbuf);
    }
    softmax_kernel<<<(Gg * Hhead * 256) / 8, 256, 0, stream>>>(Sbuf, Pbuf);
    {
      dim3 grid(2, Gg * Hhead);
      attn_pv_kernel<<<grid, 256, 0, stream>>>(Pbuf, QKV, T2);
    }
    cvt(T2, TB1, NC);
    gemm_f(TB1, 256, WB + OFF_AOUT + (size_t)l * 65536, 256, T3, 256,
           attn_out_b + l * 256, Nn, 256, 256);
    // h_attn = LN(o + h)
    ln_kernel<<<Nn / 8, 256, 0, stream>>>(T3, Hbuf, ln2_g + l * 256,
                                          ln2_b + l * 256, T3, nullptr, 0);

    // ---- combine + FFN ----
    add2_kernel<<<(unsigned)(NC / 256), 256, 0, stream>>>(T4, T3, T2, TB1, NC);
    gemm_br(TB1, 256, WB + OFF_M1 + (size_t)l * 131072, 512, TB2, 512,
            mlp_b1 + l * 512, Nn, 512, 256);
    gemm_f(TB2, 512, WB + OFF_M2 + (size_t)l * 131072, 256, T3, 256,
           mlp_b2 + l * 256, Nn, 256, 512);
    // h = LN(out + m)
    ln_kernel<<<Nn / 8, 256, 0, stream>>>(T3, T2, ln3_g + l * 256,
                                          ln3_b + l * 256, Hbuf, HB, 0);
  }

  // ---- mean pool + classifier ----
  pool_kernel<<<Gg, 256, 0, stream>>>(Hbuf, GEMB, GEMBB);
  gemm_f(GEMBB, 256, WB + OFF_C1, 256, Z1, 256, cls_b1, Gg, 256, 256);
  ln_kernel<<<Gg / 8, 256, 0, stream>>>(Z1, nullptr, cls_ln_g, cls_ln_b,
                                        nullptr, Z1B, 1);
  gemm_br(Z1B, 256, WB + OFF_C2, 128, Z2B, 128, cls_b2, Gg, 128, 256);
  cls_final_kernel<<<1, 256, 0, stream>>>(Z2B, cls_w3, cls_b3, (float*)d_out);
  (void)in_sizes; (void)n_in; (void)out_size; (void)ws_size;
}
